// MFM_48034914238543
// MI455X (gfx1250) — matrix operand order repaired and checked by reading, not compiled, not run on hardware
//
#include <hip/hip_runtime.h>
#include <hip/hip_bf16.h>

#define CEILDIV(a,b) (((a)+(b)-1)/(b))

static constexpr int  kB   = 4;
static constexpr long long kL  = 16384;            // H*W
static constexpr long long FBS = 128LL * kL;       // full (128-ch) per-batch stride
static constexpr long long HBS = 64LL  * kL;       // half (64-ch) per-batch stride
static constexpr float kBNINV  = 0.9999950000374996f; // 1/sqrt(1+1e-5)

enum { ACT_NONE = 0, ACT_RELU, ACT_SILU, ACT_SIG, ACT_GELU, ACT_ELU1 };

typedef __attribute__((ext_vector_type(16))) __bf16 v16bf;
typedef __attribute__((ext_vector_type(8)))  float  v8f;

union FragU { v16bf v; uint4 q[2]; };

__device__ __forceinline__ unsigned short f2bf(float f) {
  union { float f; unsigned u; } x; x.f = f;
  unsigned r = x.u + 0x7FFFu + ((x.u >> 16) & 1u);   // round-to-nearest-even
  return (unsigned short)(r >> 16);
}

__device__ __forceinline__ unsigned pk2bf(float a, float b) {
  return (unsigned)f2bf(a) | ((unsigned)f2bf(b) << 16);
}

__device__ __forceinline__ float act_apply(float v, int act) {
  switch (act) {
    case ACT_RELU: return fmaxf(v, 0.f);
    case ACT_SILU: return v / (1.f + __expf(-v));
    case ACT_SIG:  return 1.f / (1.f + __expf(-v));
    case ACT_GELU: return 0.5f * v * (1.f + erff(v * 0.70710678118f));
    case ACT_ELU1: return v > 0.f ? v + 1.f : __expf(v);
  }
  return v;
}

// ------------------------- Tensor Data Mover helpers ------------------------
#if __has_builtin(__builtin_amdgcn_tensor_load_to_lds)
#define HAVE_TDM 1
typedef __attribute__((ext_vector_type(4))) unsigned tdm_v4u;
typedef __attribute__((ext_vector_type(8))) int      tdm_v8i;
typedef __attribute__((ext_vector_type(4))) int      tdm_v4i;

// 2-D tile DMA: global (width x nrows, row stride in elements, 4B data) -> LDS
__device__ __forceinline__ void tdm_load_2d(const void* gaddr, unsigned lds_byte_addr,
                                            int width, int nrows, int row_stride)
{
  unsigned long long ga = (unsigned long long)gaddr;
  tdm_v4u g0;
  g0.x = 1u;                                   // count=1 (valid), user mode
  g0.y = lds_byte_addr;                        // D#.lds_addr (bytes)
  g0.z = (unsigned)(ga & 0xFFFFFFFFu);         // global_addr[31:0]  (bits 95:64)
  g0.w = (unsigned)((ga >> 32) & 0x01FFFFFFu)  // global_addr[56:32]
       | (2u << 30);                           // type = 2 ("image")
  tdm_v8i g1;
  g1[0] = 0x00020000;                                    // data_size=4B, mask=0
  g1[1] = (width & 0xFFFF) << 16;                        // tensor_dim0[15:0]
  g1[2] = ((width >> 16) & 0xFFFF) | ((nrows & 0xFFFF) << 16); // dim0 hi | dim1 lo
  g1[3] = ((nrows >> 16) & 0xFFFF) | ((width & 0xFFFF) << 16); // dim1 hi | tile_dim0
  g1[4] = (nrows & 0xFFFF);                              // tile_dim1 (tile_dim2=0)
  g1[5] = row_stride;                                    // tensor_dim0_stride[31:0]
  g1[6] = 0;
  g1[7] = 0;
  tdm_v4i gz = {0, 0, 0, 0};
#if defined(__clang_major__) && (__clang_major__ >= 23)
  tdm_v8i gz8 = {0, 0, 0, 0, 0, 0, 0, 0};
  __builtin_amdgcn_tensor_load_to_lds(g0, g1, gz, gz, gz8, 0);
#else
  __builtin_amdgcn_tensor_load_to_lds(g0, g1, gz, gz, 0);
#endif
}
#else
#define HAVE_TDM 0
#endif

// ---------------------------------------------------------------------------
// Batched GEMM:  D = act( alpha*(A@B) + beta*D + bias[m], bn(scale,shift) ) + res
// A: [M,K] row-major (per-batch stride sA; 0 broadcasts).
// B: [K,N] row-major if tb==0, else stored [N,K].
// Block: 64x64 tile, 128 threads = 4 waves, each wave 2x2 WMMA, K-step 64.
// ---------------------------------------------------------------------------
__global__ __launch_bounds__(128)
void k_gemm(const float* __restrict__ Ag, int lda, long long sA,
            const float* __restrict__ Bg, int ldb, long long sB, int tb,
            float* __restrict__ Dg, int ldd, long long sD,
            int M, int N, int K, float alpha, float beta,
            const float* __restrict__ bias,
            const float* __restrict__ scale, const float* __restrict__ shift,
            float smul, const float* __restrict__ res, int act)
{
  constexpr int AP = 72;  // LDS pitch (bf16): 144B rows, 16B-aligned fragments
  __shared__ __align__(16) unsigned short As[64 * AP];
  __shared__ __align__(16) unsigned short Bs[64 * AP];
  const int tid  = threadIdx.x;
  const int wave = tid >> 5;
  const int lane = tid & 31;
  const int ln16 = lane & 15;
  const int lh   = lane >> 4;
  const int tM = blockIdx.y * 64;
  const int tN = blockIdx.x * 64;
  const int wm = (wave >> 1) * 32;
  const int wn = (wave & 1) * 32;
  const float* Ab = Ag + (long long)blockIdx.z * sA;
  const float* Bb = Bg + (long long)blockIdx.z * sB;
  float*       Db = Dg + (long long)blockIdx.z * sD;
  const float* Rb = res ? res + (long long)blockIdx.z * sD : (const float*)0;

  const bool fast = (tM + 64 <= M) && (tN + 64 <= N) && ((K & 63) == 0) &&
                    ((lda & 3) == 0) && ((ldb & 3) == 0);

  v8f zero = {0.f,0.f,0.f,0.f,0.f,0.f,0.f,0.f};
  v8f acc[2][2];
  acc[0][0] = zero; acc[0][1] = zero; acc[1][0] = zero; acc[1][1] = zero;

  for (int kk = 0; kk < K; kk += 64) {
    __syncthreads();
    if (fast) {
      #pragma unroll
      for (int i = 0; i < 8; ++i) {
        int u = tid + i * 128;
        int r = u >> 4, kq = (u & 15) << 2;
        const float4 v = *(const float4*)(Ab + (long long)(tM + r) * lda + kk + kq);
        uint2 pk; pk.x = pk2bf(v.x, v.y); pk.y = pk2bf(v.z, v.w);
        *(uint2*)(As + r * AP + kq) = pk;
      }
      if (!tb) {
        #pragma unroll
        for (int i = 0; i < 8; ++i) {
          int u = tid + i * 128;
          int n4 = (u & 15) << 2, k = u >> 4;
          const float4 v = *(const float4*)(Bb + (long long)(kk + k) * ldb + tN + n4);
          Bs[(n4 + 0) * AP + k] = f2bf(v.x);
          Bs[(n4 + 1) * AP + k] = f2bf(v.y);
          Bs[(n4 + 2) * AP + k] = f2bf(v.z);
          Bs[(n4 + 3) * AP + k] = f2bf(v.w);
        }
      } else {
        #pragma unroll
        for (int i = 0; i < 8; ++i) {
          int u = tid + i * 128;
          int kq = (u & 15) << 2, n = u >> 4;
          const float4 v = *(const float4*)(Bb + (long long)(tN + n) * ldb + kk + kq);
          uint2 pk; pk.x = pk2bf(v.x, v.y); pk.y = pk2bf(v.z, v.w);
          *(uint2*)(Bs + n * AP + kq) = pk;
        }
      }
    } else {
      for (int e = tid; e < 4096; e += 128) {
        int r = e >> 6, k = e & 63;
        int gr = tM + r, gk = kk + k;
        int cr = gr < M ? gr : M - 1;
        int ck = gk < K ? gk : 0;
        float v = Ab[(long long)cr * lda + ck];
        v = (gr < M && gk < K) ? v : 0.f;
        As[r * AP + k] = f2bf(v);
      }
      if (!tb) {
        for (int e = tid; e < 4096; e += 128) {
          int n = e & 63, k = e >> 6;
          int gn = tN + n, gk = kk + k;
          int cn = gn < N ? gn : N - 1;
          int ck = gk < K ? gk : 0;
          float v = Bb[(long long)ck * ldb + cn];
          v = (gn < N && gk < K) ? v : 0.f;
          Bs[n * AP + k] = f2bf(v);
        }
      } else {
        for (int e = tid; e < 4096; e += 128) {
          int k = e & 63, n = e >> 6;
          int gn = tN + n, gk = kk + k;
          int cn = gn < N ? gn : N - 1;
          int ck = gk < K ? gk : 0;
          float v = Bb[(long long)cn * ldb + ck];
          v = (gn < N && gk < K) ? v : 0.f;
          Bs[n * AP + k] = f2bf(v);
        }
      }
    }
    __syncthreads();
    if (fast && (kk + 64 < K)) {
      int r = tid >> 1;
      int kq = (tid & 1) << 5;
      __builtin_prefetch(Ab + (long long)(tM + r) * lda + kk + 64 + kq, 0, 0);
    }
    #pragma unroll
    for (int ks = 0; ks < 2; ++ks) {
      const int k0 = ks << 5;
      FragU fa[2], fb[2];
      #pragma unroll
      for (int i = 0; i < 2; ++i) {
        const unsigned short* p = As + (wm + i * 16 + ln16) * AP + k0 + lh * 8;
        fa[i].q[0] = *(const uint4*)p;
        fa[i].q[1] = *(const uint4*)(p + 16);
      }
      #pragma unroll
      for (int j = 0; j < 2; ++j) {
        const unsigned short* p = Bs + (wn + j * 16 + ln16) * AP + k0 + lh * 8;
        fb[j].q[0] = *(const uint4*)p;
        fb[j].q[1] = *(const uint4*)(p + 16);
      }
      #pragma unroll
      for (int i = 0; i < 2; ++i)
        #pragma unroll
        for (int j = 0; j < 2; ++j)
          acc[i][j] = __builtin_amdgcn_wmma_f32_16x16x32_bf16(
              false, fa[i].v, false, fb[j].v, (short)0, acc[i][j], false, false);
    }
  }

  const bool efull = (tM + 64 <= M) && (tN + 64 <= N);
  #pragma unroll
  for (int i = 0; i < 2; ++i) {
    #pragma unroll
    for (int j = 0; j < 2; ++j) {
      int gn = tN + wn + j * 16 + ln16;
      if (!efull && gn >= N) continue;
      #pragma unroll
      for (int r = 0; r < 8; ++r) {
        int gm = tM + wm + i * 16 + lh * 8 + r;
        if (!efull && gm >= M) continue;
        long long idx = (long long)gm * ldd + gn;
        float v = acc[i][j][r] * alpha;
        if (beta != 0.f) v += beta * Db[idx];
        if (bias)  v += bias[gm];
        if (scale) v = v * (scale[gm] * smul) + shift[gm];
        v = act_apply(v, act);
        if (Rb) v += Rb[idx];
        Db[idx] = v;
      }
    }
  }
}

// --------------------------- elementwise / small kernels --------------------

__global__ void k_dftinit(float* __restrict__ Cm, float* __restrict__ Sm) {
  int i = blockIdx.x * blockDim.x + threadIdx.x;
  if (i >= 16384) return;
  int j = i >> 7, k = i & 127;
  int t = (j * k) & 127;                       // exact twiddle reduction
  float ang = 6.283185307179586f * (float)t * (1.f / 128.f);
  Cm[i] = cosf(ang);
  Sm[i] = sinf(ang);
}

// Depthwise / grouped conv with TDM-staged LDS tile.
// Block = (y-band of 16 rows) x (out channel) x (batch); 256 threads.
__global__ __launch_bounds__(256)
void k_dwconv2(const float* __restrict__ in, const float* __restrict__ wt,
               const float* __restrict__ bs, float* __restrict__ out,
               int Cin, int Cout, int cinper, int ks, int pad,
               int accum, int act)
{
  __shared__ float tile[2][20 * 128];          // up to (16+2*pad) x 128, 2 in-ch
  const int tid = threadIdx.x;
  const int y0  = blockIdx.x * 16;
  const int co  = blockIdx.y;
  const int b   = blockIdx.z;
  const int rows_t = 16 + 2 * pad;
  int ys = y0 - pad; if (ys < 0) ys = 0;
  int ye = y0 + 16 + pad; if (ye > 128) ye = 128;
  const int nrows = ye - ys;
  const int top = ys - (y0 - pad);             // missing halo rows above
  const int bot = rows_t - top - nrows;        // missing halo rows below
  const int ci0 = co * cinper;

  // zero out-of-image halo rows (disjoint from the DMA target region)
  for (int ci = 0; ci < cinper; ++ci) {
    for (int e = tid; e < top * 128; e += 256) tile[ci][e] = 0.f;
    for (int e = tid; e < bot * 128; e += 256) tile[ci][(top + nrows) * 128 + e] = 0.f;
  }
#if HAVE_TDM
  if (tid < 32) {                              // wave 0 issues the tensor DMA
    for (int ci = 0; ci < cinper; ++ci) {
      const float* src = in + (((long long)b * Cin + ci0 + ci) << 14) + ys * 128;
      unsigned laddr = (unsigned)(unsigned long long)&tile[ci][top * 128];
      tdm_load_2d(src, laddr, 128, nrows, 128);
    }
    __builtin_amdgcn_s_wait_tensorcnt(0);
  }
#else
  for (int ci = 0; ci < cinper; ++ci) {
    const float* src = in + (((long long)b * Cin + ci0 + ci) << 14) + ys * 128;
    for (int e = tid; e < nrows * 128; e += 256)
      tile[ci][top * 128 + e] = src[e];
  }
#endif
  __syncthreads();

  #pragma unroll
  for (int p = 0; p < 8; ++p) {
    int px = tid + p * 256;                    // 16*128 pixels / 256 threads
    int row = px >> 7, col = px & 127;
    float acc = bs ? bs[co] : 0.f;
    for (int ci = 0; ci < cinper; ++ci) {
      const float* wp = wt + (long long)(co * cinper + ci) * ks * ks;  // uniform
      for (int ky = 0; ky < ks; ++ky)
        for (int kx = 0; kx < ks; ++kx) {
          int c2 = col + kx - pad;
          float v = ((unsigned)c2 < 128u) ? tile[ci][(row + ky) * 128 + c2] : 0.f;
          acc += v * wp[ky * ks + kx];
        }
    }
    acc = act_apply(acc, act);
    long long oi = (((long long)b * Cout + co) << 14) + ((y0 + row) << 7) + col;
    if (accum) out[oi] += acc; else out[oi] = acc;
  }
}

__global__ void k_ln(const float* __restrict__ in, float* __restrict__ out,
                     const float* __restrict__ g, const float* __restrict__ bb, int C)
{
  int i = blockIdx.x * blockDim.x + threadIdx.x;
  if (i >= kB * 16384) return;
  int b = i >> 14, l = i & 16383;
  const float* p = in + (((long long)b * C) << 14) + l;
  float s = 0.f;
  for (int c = 0; c < C; ++c) s += p[(long long)c << 14];
  float m = s / C;
  float v = 0.f;
  for (int c = 0; c < C; ++c) { float d = p[(long long)c << 14] - m; v += d * d; }
  float r = rsqrtf(v / C + 1e-5f);
  float* q = out + (((long long)b * C) << 14) + l;
  for (int c = 0; c < C; ++c)
    q[(long long)c << 14] = (p[(long long)c << 14] - m) * r * g[c] + bb[c];
}

__global__ void k_rope(const float* __restrict__ in, long long ibs,
                       float* __restrict__ out, long long obs)
{
  int i = blockIdx.x * blockDim.x + threadIdx.x;
  if (i >= kB * 32 * 16384) return;
  int l = i & 16383;
  int j = (i >> 14) & 31;
  int b = i >> 19;
  int hh = l >> 7, ww = l & 127;
  int m = j & 15;
  float th = __powf(10000.f, -(float)m * (1.f / 16.f));
  float ang = ((j < 16) ? (float)hh : (float)ww) * th;
  float sn, cs; __sincosf(ang, &sn, &cs);
  const float* p = in + (long long)b * ibs + ((long long)(2 * j) << 14) + l;
  float xr = p[0], xi = p[16384];
  float* q = out + (long long)b * obs + ((long long)(2 * j) << 14) + l;
  q[0]     = cs * xr - sn * xi;
  q[16384] = sn * xr + cs * xi;
}

__global__ void k_colmean(const float* __restrict__ in, long long bs, float* __restrict__ om)
{
  int bc = blockIdx.x; int b = bc >> 6, c = bc & 63;
  const float* p = in + (long long)b * bs + ((long long)c << 14);
  __shared__ float sm[256];
  float s = 0.f;
  for (int l = threadIdx.x; l < 16384; l += 256) s += p[l];
  sm[threadIdx.x] = s; __syncthreads();
  for (int st2 = 128; st2 > 0; st2 >>= 1) {
    if ((int)threadIdx.x < st2) sm[threadIdx.x] += sm[threadIdx.x + st2];
    __syncthreads();
  }
  if (threadIdx.x == 0) om[bc] = sm[0] * (1.f / 16384.f);
}

__global__ void k_z(const float* __restrict__ q, long long qbs,
                    const float* __restrict__ km, float* __restrict__ z)
{
  int i = blockIdx.x * blockDim.x + threadIdx.x;
  if (i >= kB * 8 * 16384) return;
  int l = i & 16383; int h = (i >> 14) & 7; int b = i >> 17;
  const float* qp = q + (long long)b * qbs + ((long long)(h * 8) << 14) + l;
  const float* kp = km + b * 64 + h * 8;
  float s = 1e-6f;
  #pragma unroll
  for (int d = 0; d < 8; ++d) s += qp[(long long)d << 14] * kp[d];
  z[i] = 1.f / s;
}

__global__ void k_kv(const float* __restrict__ kr, long long kbs,
                     const float* __restrict__ v, long long vbs,
                     float* __restrict__ kv)
{
  int bh = blockIdx.x; int b = bh >> 3, h = bh & 7;
  int pair = threadIdx.x & 63;
  int grp  = threadIdx.x >> 6;
  int d = pair >> 3, e = pair & 7;
  const float* kp = kr + (long long)b * kbs + ((long long)(h * 8 + d) << 14);
  const float* vp = v  + (long long)b * vbs + ((long long)(h * 8 + e) << 14);
  float s = 0.f;
  for (int l = grp * 4096; l < (grp + 1) * 4096; ++l) s += kp[l] * vp[l];
  __shared__ float sm[256];
  sm[threadIdx.x] = s;
  __syncthreads();
  if (threadIdx.x < 64) {
    float t = sm[pair] + sm[pair + 64] + sm[pair + 128] + sm[pair + 192];
    kv[bh * 64 + pair] = t * (1.f / 16384.f);   // s^2 = 1/n
  }
}

__global__ void k_attnout(const float* __restrict__ qr, long long qbs,
                          const float* __restrict__ kv, const float* __restrict__ z,
                          float* __restrict__ out, long long obs)
{
  int bh = blockIdx.y; int b = bh >> 3, h = bh & 7;
  __shared__ float skv[64];
  if (threadIdx.x < 64) skv[threadIdx.x] = kv[bh * 64 + threadIdx.x];
  __syncthreads();
  int l = blockIdx.x * 256 + threadIdx.x;
  const float* qp = qr + (long long)b * qbs + ((long long)(h * 8) << 14) + l;
  float qv[8];
  #pragma unroll
  for (int d = 0; d < 8; ++d) qv[d] = qp[(long long)d << 14];
  float zz = z[((long long)bh << 14) + l];
  float* op = out + (long long)b * obs + ((long long)(h * 8) << 14) + l;
  #pragma unroll
  for (int e = 0; e < 8; ++e) {
    float s = 0.f;
    #pragma unroll
    for (int d = 0; d < 8; ++d) s += qv[d] * skv[d * 8 + e];
    op[(long long)e << 14] = s * zz;
  }
}

__global__ void k_add3(float* __restrict__ o, const float* __restrict__ a,
                       const float* __restrict__ b2, const float* __restrict__ c, int n)
{ int i = blockIdx.x * blockDim.x + threadIdx.x; if (i < n) o[i] = a[i] + b2[i] + c[i]; }

__global__ void k_modmul(float* __restrict__ zr, float* __restrict__ zi,
                         const float* __restrict__ w, const float* __restrict__ tr,
                         const float* __restrict__ ti, int n)
{
  int i = blockIdx.x * blockDim.x + threadIdx.x;
  if (i < n) { float s = w[i]; zr[i] = s * tr[i]; zi[i] = s * ti[i]; }
}

__global__ void k_absbnrelu(float* __restrict__ out, const float* __restrict__ re,
                            const float* __restrict__ im, const float* __restrict__ g,
                            const float* __restrict__ be, float sc, int n)
{
  int i = blockIdx.x * blockDim.x + threadIdx.x;
  if (i >= n) return;
  int c = (i >> 14) & 127;
  float a = re[i], b2 = im[i];
  float v = sqrtf(a * a + b2 * b2) * sc;
  v = v * (g[c] * kBNINV) + be[c];
  out[i] = fmaxf(v, 0.f);
}

__global__ void k_concatmul(float* __restrict__ out, const float* __restrict__ a3,
                            const float* __restrict__ a5, const float* __restrict__ ar)
{
  int i = blockIdx.x * blockDim.x + threadIdx.x;
  if (i >= kB * 128 * 16384) return;
  int l = i & 16383; int c = (i >> 14) & 127; int b = i >> 21;
  float t = (c < 64) ? a3[((((long long)b << 6) | c) << 14) | l]
                     : a5[((((long long)b << 6) | (c - 64)) << 14) | l];
  out[i] = t * ar[i];
}

__global__ void k_concat(float* __restrict__ cat, const float* __restrict__ x0,
                         const float* __restrict__ xo)
{
  int i = blockIdx.x * blockDim.x + threadIdx.x;
  if (i >= kB * 256 * 16384) return;
  int l = i & 16383; int c = (i >> 14) & 255; int b = i >> 22;
  cat[i] = (c < 128) ? x0[((((long long)b << 7) | c) << 14) | l]
                     : xo[((((long long)b << 7) | (c - 128)) << 14) | l];
}

// ------------------------------- host helpers -------------------------------

static void gemm(hipStream_t st,
                 const float* A, int lda, long long sA,
                 const float* Bm, int ldb, long long sB, int tb,
                 float* D, int ldd, long long sD,
                 int M, int N, int K, int batch,
                 float alpha, float beta,
                 const float* bias, const float* scale, const float* shift, float smul,
                 const float* res, int act)
{
  dim3 grid(CEILDIV(N, 64), CEILDIV(M, 64), batch);
  k_gemm<<<grid, dim3(128), 0, st>>>(A, lda, sA, Bm, ldb, sB, tb, D, ldd, sD,
                                     M, N, K, alpha, beta, bias, scale, shift, smul, res, act);
}

static void dwconv(hipStream_t st, const float* in, const float* w, const float* b,
                   float* out, int Cin, int Cout, int cinper, int ks, int pad,
                   int accum, int act)
{
  dim3 grid(8, Cout, kB);
  k_dwconv2<<<grid, 256, 0, st>>>(in, w, b, out, Cin, Cout, cinper, ks, pad, accum, act);
}

// fmt_block: fft2 / weighting / ifft2 via DFT matmuls (F = C - iS, symmetric)
static void fmt_run(hipStream_t st, const float* X, float* out,
                    float* tr, float* ti, float* u1, float* u2, float* wgt, float* w8,
                    const float* Cm, const float* Sm,
                    const float* fw1, const float* fb1, const float* f_g, const float* f_be,
                    const float* fw2, const float* fb2, const float* nbn_g, const float* nbn_be)
{
  const int NL = (int)kL;
  const int bat = kB * 128;   // 512 independent 128x128 images
  gemm(st, Cm,128,0,  X,128,16384,0, u1,128,16384, 128,128,128,bat, 1.f,0.f, 0,0,0,1.f,0,ACT_NONE);
  gemm(st, Sm,128,0,  X,128,16384,0, u2,128,16384, 128,128,128,bat, 1.f,0.f, 0,0,0,1.f,0,ACT_NONE);
  gemm(st, u1,128,16384, Cm,128,0,0, tr,128,16384, 128,128,128,bat, 1.f,0.f, 0,0,0,1.f,0,ACT_NONE);
  gemm(st, u2,128,16384, Sm,128,0,0, tr,128,16384, 128,128,128,bat,-1.f,1.f, 0,0,0,1.f,0,ACT_NONE);
  gemm(st, u1,128,16384, Sm,128,0,0, ti,128,16384, 128,128,128,bat,-1.f,0.f, 0,0,0,1.f,0,ACT_NONE);
  gemm(st, u2,128,16384, Cm,128,0,0, ti,128,16384, 128,128,128,bat,-1.f,1.f, 0,0,0,1.f,0,ACT_NONE);
  gemm(st, fw1,128,0, tr,NL,FBS,0, w8,NL,8*kL, 8,NL,128,kB, 1.f,0.f, fb1,f_g,f_be,kBNINV, 0,ACT_RELU);
  gemm(st, fw2,8,0,   w8,NL,8*kL,0, wgt,NL,FBS, 128,NL,8,kB, 1.f,0.f, fb2,0,0,1.f, 0,ACT_SIG);
  { int n = (int)(kB * FBS);
    k_modmul<<<CEILDIV(n,256),256,0,st>>>(u1, u2, wgt, tr, ti, n); }
  gemm(st, Cm,128,0, u1,128,16384,0, tr,128,16384, 128,128,128,bat, 1.f,0.f, 0,0,0,1.f,0,ACT_NONE);
  gemm(st, Sm,128,0, u2,128,16384,0, tr,128,16384, 128,128,128,bat,-1.f,1.f, 0,0,0,1.f,0,ACT_NONE);
  gemm(st, Cm,128,0, u2,128,16384,0, ti,128,16384, 128,128,128,bat, 1.f,0.f, 0,0,0,1.f,0,ACT_NONE);
  gemm(st, Sm,128,0, u1,128,16384,0, ti,128,16384, 128,128,128,bat, 1.f,1.f, 0,0,0,1.f,0,ACT_NONE);
  gemm(st, tr,128,16384, Cm,128,0,0, u1,128,16384, 128,128,128,bat, 1.f,0.f, 0,0,0,1.f,0,ACT_NONE);
  gemm(st, ti,128,16384, Sm,128,0,0, u1,128,16384, 128,128,128,bat,-1.f,1.f, 0,0,0,1.f,0,ACT_NONE);
  gemm(st, tr,128,16384, Sm,128,0,0, u2,128,16384, 128,128,128,bat, 1.f,0.f, 0,0,0,1.f,0,ACT_NONE);
  gemm(st, ti,128,16384, Cm,128,0,0, u2,128,16384, 128,128,128,bat, 1.f,1.f, 0,0,0,1.f,0,ACT_NONE);
  { int n = (int)(kB * FBS);
    k_absbnrelu<<<CEILDIV(n,256),256,0,st>>>(out, u1, u2, nbn_g, nbn_be, 1.f/16384.f, n); }
}

static void attn_run(hipStream_t st, const float* t, float* out, float* qkbuf,
                     float* qr, float* kr, float* zb, float* kmb, float* kvb,
                     const float* qk_w, const float* qk_b,
                     const float* lepe_w, const float* lepe_b)
{
  const int NL = (int)kL;
  gemm(st, qk_w,64,0, t,NL,HBS,0, qkbuf,NL,FBS, 128,NL,64,kB, 1.f,0.f, qk_b,0,0,1.f, 0,ACT_ELU1);
  int nr = kB * 32 * NL;
  k_rope<<<CEILDIV(nr,256),256,0,st>>>(qkbuf,           FBS, qr, HBS);
  k_rope<<<CEILDIV(nr,256),256,0,st>>>(qkbuf + 64 * kL, FBS, kr, HBS);
  k_colmean<<<kB*64, 256, 0, st>>>(qkbuf + 64 * kL, FBS, kmb);
  int nz = kB * 8 * NL;
  k_z<<<CEILDIV(nz,256),256,0,st>>>(qkbuf, FBS, kmb, zb);
  k_kv<<<kB*8, 256, 0, st>>>(kr, HBS, t, HBS, kvb);
  dim3 gao(NL / 256, kB * 8);
  k_attnout<<<gao, 256, 0, st>>>(qr, HBS, kvb, zb, out, HBS);
  dwconv(st, t, lepe_w, lepe_b, out, 64, 64, 1, 3, 1, 1, ACT_NONE);
}

// --------------------------------- driver -----------------------------------

extern "C" void kernel_launch(void* const* d_in, const int* in_sizes, int n_in,
                              void* d_out, int out_size, void* d_ws, size_t ws_size,
                              hipStream_t stream)
{
  (void)in_sizes; (void)n_in; (void)out_size; (void)ws_size;
  const float* x      = (const float*)d_in[0];
  const float* c1_w   = (const float*)d_in[1];
  const float* c1_b   = (const float*)d_in[2];
  const float* c1_g   = (const float*)d_in[3];
  const float* c1_be  = (const float*)d_in[4];
  const float* cpe1_w = (const float*)d_in[5];
  const float* cpe1_b = (const float*)d_in[6];
  const float* fw1    = (const float*)d_in[7];
  const float* fb1    = (const float*)d_in[8];
  const float* f_g    = (const float*)d_in[9];
  const float* f_be   = (const float*)d_in[10];
  const float* fw2    = (const float*)d_in[11];
  const float* fb2    = (const float*)d_in[12];
  const float* nbn_g  = (const float*)d_in[13];
  const float* nbn_be = (const float*)d_in[14];
  const float* ln1_g  = (const float*)d_in[15];
  const float* ln1_b  = (const float*)d_in[16];
  const float* d3w1   = (const float*)d_in[17];
  const float* d3b1   = (const float*)d_in[18];
  const float* d3w2   = (const float*)d_in[19];
  const float* d5w1   = (const float*)d_in[20];
  const float* d5b1   = (const float*)d_in[21];
  const float* d5w2   = (const float*)d_in[22];
  const float* ap_w   = (const float*)d_in[23];
  const float* ap_b   = (const float*)d_in[24];
  const float* ip2_w  = (const float*)d_in[25];
  const float* ip2_b  = (const float*)d_in[26];
  const float* dwc2_w = (const float*)d_in[27];
  const float* dwc2_b = (const float*)d_in[28];
  const float* qk_w   = (const float*)d_in[29];
  const float* qk_b   = (const float*)d_in[30];
  const float* lepe_w = (const float*)d_in[31];
  const float* lepe_b = (const float*)d_in[32];
  const float* op_w   = (const float*)d_in[33];
  const float* op_b   = (const float*)d_in[34];
  const float* cpe2_w = (const float*)d_in[35];
  const float* cpe2_b = (const float*)d_in[36];
  const float* ln2_g  = (const float*)d_in[37];
  const float* ln2_b  = (const float*)d_in[38];
  const float* ffn_w1 = (const float*)d_in[39];
  const float* ffn_b1 = (const float*)d_in[40];
  const float* ffn_w2 = (const float*)d_in[41];
  const float* ffn_b2 = (const float*)d_in[42];
  const float* po_w   = (const float*)d_in[43];
  const float* red_w  = (const float*)d_in[44];
  const float* red_b  = (const float*)d_in[45];
  const float* red_g  = (const float*)d_in[46];
  const float* red_be = (const float*)d_in[47];

  float* wsf = (float*)d_ws;
  long long off = 0;
  auto alloc = [&](long long n) { float* p = wsf + off; off += n; return p; };
  const long long BIGN  = kB * FBS;   // 8388608 floats
  const long long HALFN = kB * HBS;

  float* Cm  = alloc(16384);
  float* Sm  = alloc(16384);
  float* x0  = alloc(BIGN);
  float* xt  = alloc(BIGN);   // shortcut; later fmt2
  float* bP2 = alloc(BIGN);   // fmt1 -> xt2 -> xt3
  float* s0  = alloc(BIGN);   // scratch (fmt tr / qk / cat lo)
  float* s1  = alloc(BIGN);   // scratch (fmt ti / xs2 / cat hi)
  float* s2  = alloc(BIGN);   // fmt u1
  float* s3  = alloc(BIGN);   // fmt u2
  float* s4  = alloc(BIGN);   // fmt wgt
  float* bP8 = alloc(BIGN);   // x_s -> h2 -> xo
  float* bPF = alloc(BIGN);   // act_res -> ffn_out
  float* h35 = alloc(HALFN);  // x_s3 -> a3
  float* h55 = alloc(HALFN);  // x_s5 -> a5
  float* t3  = alloc(HALFN);
  float* t5  = alloc(HALFN);
  float* qr  = alloc(HALFN);
  float* kr  = alloc(HALFN);
  float* g1  = alloc(kB * 4 * kL);
  float* zb  = alloc(kB * 8 * kL);   // z; reused as fmt w8
  float* kmb = alloc(kB * 64);
  float* kvb = alloc(kB * 8 * 8 * 8);

  const int NL = (int)kL;

  k_dftinit<<<64, 256, 0, stream>>>(Cm, Sm);

  // x_0 = relu(bn(conv1x1(x, c1)))
  gemm(stream, c1_w,128,0, x,NL,FBS,0, x0,NL,FBS, 128,NL,128,kB,
       1.f,0.f, c1_b, c1_g, c1_be, kBNINV, nullptr, ACT_RELU);

  // xt = x + cpe1(x)
  hipMemcpyAsync(xt, x, BIGN * sizeof(float), hipMemcpyDeviceToDevice, stream);
  dwconv(stream, x, cpe1_w, cpe1_b, xt, 128, 128, 1, 3, 1, 1, ACT_NONE);

  // fmt1 -> bP2
  fmt_run(stream, xt, bP2, s0, s1, s2, s3, s4, zb, Cm, Sm,
          fw1, fb1, f_g, f_be, fw2, fb2, nbn_g, nbn_be);

  // x_s = ln(xt)
  { int n = kB * NL; k_ln<<<CEILDIV(n,256),256,0,stream>>>(xt, bP8, ln1_g, ln1_b, 128); }

  // x_s3 / x_s5 (1x1 + grouped conv, 2 in-ch per group)
  gemm(stream, d3w1,128,0, bP8,NL,FBS,0, s0,NL,FBS, 128,NL,128,kB, 1.f,0.f, d3b1,0,0,1.f, 0,ACT_NONE);
  dwconv(stream, s0, d3w2, nullptr, h35, 128, 64, 2, 3, 1, 0, ACT_NONE);
  gemm(stream, d5w1,128,0, bP8,NL,FBS,0, s0,NL,FBS, 128,NL,128,kB, 1.f,0.f, d5b1,0,0,1.f, 0,ACT_NONE);
  dwconv(stream, s0, d5w2, nullptr, h55, 128, 64, 2, 5, 2, 0, ACT_NONE);

  // act_res = silu(ap(x_s))
  gemm(stream, ap_w,128,0, bP8,NL,FBS,0, bPF,NL,FBS, 128,NL,128,kB, 1.f,0.f, ap_b,0,0,1.f, 0,ACT_SILU);

  // branches: t = silu(dwc2(ip2(x_s*)))
  gemm(stream, ip2_w,64,0, h35,NL,HBS,0, s1,NL,HBS, 64,NL,64,kB, 1.f,0.f, ip2_b,0,0,1.f, 0,ACT_NONE);
  dwconv(stream, s1, dwc2_w, dwc2_b, t3, 64, 64, 1, 3, 1, 0, ACT_SILU);
  gemm(stream, ip2_w,64,0, h55,NL,HBS,0, s1,NL,HBS, 64,NL,64,kB, 1.f,0.f, ip2_b,0,0,1.f, 0,ACT_NONE);
  dwconv(stream, s1, dwc2_w, dwc2_b, t5, 64, 64, 1, 3, 1, 0, ACT_SILU);

  // linear attention (a3 reuses x_s3 buffer, a5 reuses x_s5 buffer)
  attn_run(stream, t3, h35, s0, qr, kr, zb, kmb, kvb, qk_w, qk_b, lepe_w, lepe_b);
  attn_run(stream, t5, h55, s0, qr, kr, zb, kmb, kvb, qk_w, qk_b, lepe_w, lepe_b);

  // xs = op( concat(a3,a5) * act_res )
  { int n = (int)BIGN; k_concatmul<<<CEILDIV(n,256),256,0,stream>>>(s0, h35, h55, bPF); }
  gemm(stream, op_w,128,0, s0,NL,FBS,0, s1,NL,FBS, 128,NL,128,kB, 1.f,0.f, op_b,0,0,1.f, 0,ACT_NONE);

  // xt2 = shortcut + xs + fmt1   (alias-safe elementwise)
  { int n = (int)BIGN; k_add3<<<CEILDIV(n,256),256,0,stream>>>(bP2, xt, s1, bP2, n); }

  // xt2 += cpe2(xt2)
  hipMemcpyAsync(s0, bP2, BIGN * sizeof(float), hipMemcpyDeviceToDevice, stream);
  dwconv(stream, s0, cpe2_w, cpe2_b, bP2, 128, 128, 1, 3, 1, 1, ACT_NONE);

  // fmt2 -> xt buffer
  fmt_run(stream, bP2, xt, s0, s1, s2, s3, s4, zb, Cm, Sm,
          fw1, fb1, f_g, f_be, fw2, fb2, nbn_g, nbn_be);

  // h2 = ln(xt2); ffn = gelu(h2 @ w1.T + b1) @ w2.T + b2
  { int n = kB * NL; k_ln<<<CEILDIV(n,256),256,0,stream>>>(bP2, bP8, ln2_g, ln2_b, 128); }
  gemm(stream, ffn_w1,128,0, bP8,NL,FBS,0, g1,NL,4*kL, 4,NL,128,kB, 1.f,0.f, ffn_b1,0,0,1.f, 0,ACT_GELU);
  gemm(stream, ffn_w2,4,0,   g1,NL,4*kL,0, bPF,NL,FBS, 128,NL,4,kB, 1.f,0.f, ffn_b2,0,0,1.f, 0,ACT_NONE);

  // xt3 = xt2 + ffn + fmt2
  { int n = (int)BIGN; k_add3<<<CEILDIV(n,256),256,0,stream>>>(bP2, bP2, bPF, xt, n); }

  // xo = po(xt3)
  gemm(stream, po_w,128,0, bP2,NL,FBS,0, bP8,NL,FBS, 128,NL,128,kB, 1.f,0.f, 0,0,0,1.f, 0,ACT_NONE);

  // cat = [x0 ; xo]  (s0,s1 are contiguous in the arena)
  { int n = (int)(2 * BIGN); k_concat<<<CEILDIV(n,256),256,0,stream>>>(s0, x0, bP8); }

  // out = relu(bn(red(cat))) + x0
  gemm(stream, red_w,256,0, s0,NL,256*kL,0, (float*)d_out,NL,FBS, 128,NL,256,kB,
       1.f,0.f, red_b, red_g, red_be, kBNINV, x0, ACT_RELU);
}
